// RelativeGlobalAttention_40991167873043
// MI455X (gfx1250) — matrix operand order repaired and checked by reading, not compiled, not run on hardware
//
#include <hip/hip_runtime.h>

#define LSEQ   2048
#define DH     64
#define NH     8
#define DMODEL 512
#define KTILES (LSEQ / 64)

typedef __attribute__((ext_vector_type(16))) _Float16 v16h;
typedef __attribute__((ext_vector_type(8)))  _Float16 v8h;
typedef __attribute__((ext_vector_type(8)))  float    v8f;
typedef __attribute__((ext_vector_type(4)))  float    v4f;
typedef __attribute__((ext_vector_type(4)))  unsigned v4ui;
typedef __attribute__((ext_vector_type(8)))  int      v8i;
typedef __attribute__((ext_vector_type(4)))  int      v4i;

#if defined(__has_builtin)
# if __has_builtin(__builtin_amdgcn_tensor_load_to_lds) && \
     __has_builtin(__builtin_amdgcn_s_wait_tensorcnt)
#  define USE_TDM 1
# endif
#endif
#ifndef USE_TDM
# define USE_TDM 0
#endif

#define WMMA_F16(a, b, c) \
  __builtin_amdgcn_wmma_f32_16x16x32_f16(false, (a), false, (b), (short)0, (c), false, false)

// A-fragment (16x32 f16), row-major source. ISA 7.12.2 lane layout:
// lanes 0-15: K=kb..kb+7, kb+16..kb+23 with kb=0; lanes 16-31: kb=8.
static __device__ __forceinline__ v16h frag_a(const _Float16* p, int stride, int row,
                                              int k0, int lane) {
  const _Float16* q = p + (long)row * stride + k0 + ((lane < 16) ? 0 : 8);
  v8h lo = *(const v8h*)(q);
  v8h hi = *(const v8h*)(q + 16);
  v16h r;
#pragma unroll
  for (int i = 0; i < 8; ++i) { r[i] = lo[i]; r[i + 8] = hi[i]; }
  return r;
}

// B-fragment (32x16 f16) from TRANSPOSED storage Bt[n][k].
// Lane n=lane&15 holds K=kb..kb+7, kb+16..kb+23, kb = (lane<16) ? 0 : 8.
static __device__ __forceinline__ v16h frag_b(const _Float16* p, int stride, int col,
                                              int k0, int lane) {
  const _Float16* q = p + (long)col * stride + k0 + ((lane < 16) ? 0 : 8);
  v8h lo = *(const v8h*)(q);
  v8h hi = *(const v8h*)(q + 16);
  v16h r;
#pragma unroll
  for (int i = 0; i < 8; ++i) { r[i] = lo[i]; r[i + 8] = hi[i]; }
  return r;
}

#if USE_TDM
// Issue one TDM 2D tile load: tile_d1 rows x tile_d0 f16 elements, row stride
// stride0 (elements), global tile start gptr -> LDS byte address lds_addr.
// D# packing per CDNA5 ISA 8.3/8.4 (group0: count/lds/global/type, group1: dims).
static __device__ __forceinline__ void tdm_load_tile_f16(unsigned lds_addr,
                                                         const _Float16* gptr,
                                                         unsigned tile_d0, unsigned tile_d1,
                                                         unsigned long stride0) {
  unsigned long ga = (unsigned long)(uintptr_t)gptr;
  v4ui g0;
  g0[0] = 1u;                                             // count=1 (valid user D#)
  g0[1] = lds_addr;                                       // bits 63:32 lds_addr
  g0[2] = (unsigned)(ga & 0xFFFFFFFFu);                   // global_addr[31:0]
  g0[3] = (unsigned)((ga >> 32) & 0x01FFFFFFu) | (2u << 30); // ga[56:32] | type=2
  v8i g1;
  g1[0] = 0x00010000;                                     // data_size=1 (2 bytes)
  g1[1] = (int)((tile_d0 & 0xFFFFu) << 16);               // tensor_dim0[15:0]
  g1[2] = (int)(((tile_d0 >> 16) & 0xFFFFu) | ((tile_d1 & 0xFFFFu) << 16)); // dim0 hi | dim1 lo
  g1[3] = (int)(((tile_d1 >> 16) & 0xFFFFu) | ((tile_d0 & 0xFFFFu) << 16)); // dim1 hi | tile_dim0
  g1[4] = (int)(tile_d1 & 0xFFFFu);                       // tile_dim1 (tile_dim2=0)
  g1[5] = (int)(stride0 & 0xFFFFFFFFu);                   // tensor_dim0_stride[31:0]
  g1[6] = (int)((stride0 >> 32) & 0xFFFFu);               // stride0[47:32], dim1_stride lo=0
  g1[7] = 0;
  v4i g2 = {0, 0, 0, 0};
  v4i g3 = {0, 0, 0, 0};
#if defined(__clang_major__) && (__clang_major__ >= 23)
  v8i g4 = {0, 0, 0, 0, 0, 0, 0, 0};
  __builtin_amdgcn_tensor_load_to_lds(g0, g1, g2, g3, g4, 0);
#else
  __builtin_amdgcn_tensor_load_to_lds(g0, g1, g2, g3, 0);
#endif
}
#endif

// ---------------------------------------------------------------------------
// Projection GEMM:  X[B*L,512] @ W[512,512] -> f16 head layouts.
//  which 0/1 (q/k): out[(b*NH+h)*L + l][d]      (row-major, dh=64)
//  which 2   (v)  : out[(b*NH+h)*64 + d][l]     (transposed for P*V B-frags)
// ---------------------------------------------------------------------------
__global__ __launch_bounds__(256) void rga_proj(
    const float* __restrict__ q, const float* __restrict__ k, const float* __restrict__ v,
    const float* __restrict__ Wq, const float* __restrict__ Wk, const float* __restrict__ Wv,
    _Float16* __restrict__ qh, _Float16* __restrict__ kh, _Float16* __restrict__ vt) {
  __shared__ _Float16 Alds[64][32];
  __shared__ _Float16 Blds[64][32];   // stored transposed: Blds[n][k]

  const int per   = gridDim.x / 3;
  const int which = blockIdx.x / per;
  const int rem   = blockIdx.x % per;
  const int mt = rem >> 3, nt = rem & 7;
  const int r0 = mt * 64, n0 = nt * 64;

  const float* X = (which == 0) ? q : (which == 1) ? k : v;
  const float* W = (which == 0) ? Wq : (which == 1) ? Wk : Wv;

  const int tid = threadIdx.x, lane = tid & 31, wave = tid >> 5;
  const int rb = wave & 3, cb0 = (wave >> 2) * 2;

  v8f c0 = {}; v8f c1 = {};

  for (int k0 = 0; k0 < DMODEL; k0 += 32) {
    __syncthreads();
    {   // stage A tile 64x32 (fp32 -> f16)
      int row = tid >> 2, col = (tid & 3) * 8;
      const float* src = X + (long)(r0 + row) * DMODEL + k0 + col;
#pragma unroll
      for (int i = 0; i < 8; i += 4) {
        v4f f = *(const v4f*)(src + i);
        Alds[row][col + i + 0] = (_Float16)f[0];
        Alds[row][col + i + 1] = (_Float16)f[1];
        Alds[row][col + i + 2] = (_Float16)f[2];
        Alds[row][col + i + 3] = (_Float16)f[3];
      }
    }
    {   // stage W tile transposed: Blds[n][kk] = W[k0+kk][n0+n]
      int kk = tid >> 3, nn = (tid & 7) * 8;
      const float* src = W + (long)(k0 + kk) * DMODEL + n0 + nn;
#pragma unroll
      for (int i = 0; i < 8; i += 4) {
        v4f f = *(const v4f*)(src + i);
        Blds[nn + i + 0][kk] = (_Float16)f[0];
        Blds[nn + i + 1][kk] = (_Float16)f[1];
        Blds[nn + i + 2][kk] = (_Float16)f[2];
        Blds[nn + i + 3][kk] = (_Float16)f[3];
      }
    }
    __syncthreads();
    v16h a  = frag_a(&Alds[0][0], 32, rb * 16 + (lane & 15), 0, lane);
    v16h b0 = frag_b(&Blds[0][0], 32, cb0 * 16 + (lane & 15), 0, lane);
    v16h b1 = frag_b(&Blds[0][0], 32, (cb0 + 1) * 16 + (lane & 15), 0, lane);
    c0 = WMMA_F16(a, b0, c0);
    c1 = WMMA_F16(a, b1, c1);
  }

  const int rof = (lane < 16) ? 0 : 8;
  const int cc  = lane & 15;
  const int h   = nt;
  _Float16* dstQK = (which == 0) ? qh : kh;
#pragma unroll
  for (int r = 0; r < 8; ++r) {
    int gr = r0 + rb * 16 + r + rof;
    int b  = gr >> 11;                      // / LSEQ
    int l  = gr & (LSEQ - 1);
    long bh = (long)b * NH + h;
    int d0 = cb0 * 16 + cc, d1 = d0 + 16;
    _Float16 x0 = (_Float16)c0[r];
    _Float16 x1 = (_Float16)c1[r];
    if (which < 2) {
      dstQK[(bh * LSEQ + l) * DH + d0] = x0;
      dstQK[(bh * LSEQ + l) * DH + d1] = x1;
    } else {
      vt[(bh * DH + d0) * LSEQ + l] = x0;
      vt[(bh * DH + d1) * LSEQ + l] = x1;
    }
  }
}

// ---------------------------------------------------------------------------
// Fused flash attention with Music-Transformer skew + TDM tile staging.
// Block = (b, h, 64-row Q tile).  256 threads = 8 wave32.
// ---------------------------------------------------------------------------
__global__ __launch_bounds__(256) void rga_attn(
    const _Float16* __restrict__ qh, const _Float16* __restrict__ kh,
    const _Float16* __restrict__ vt, const float* __restrict__ E,
    float* __restrict__ out) {
  __shared__ float    S_lds[64][65];        // QK^T tile (padded vs bank conflicts)
  __shared__ _Float16 G_lds[66][129];       // Q*Efold^T, rows l0..l0+65
  __shared__ _Float16 KP_lds[64][64];       // time-multiplexed: K tile -> P tile
  __shared__ _Float16 Ef_lds[128][64];      // folded E (B^T); rows 0..63 reused for V tile
  __shared__ float    alpha_lds[64];
  __shared__ float    scale_lds[64];

  const int tid = threadIdx.x, lane = tid & 31, wave = tid >> 5;
  const int bh = blockIdx.x >> 5;           // / (LSEQ/64)
  const int qt = blockIdx.x & 31;
  const int l0 = qt * 64;
  const int b  = bh >> 3, h = bh & 7;

  const _Float16* Q  = qh + (long)bh * LSEQ * DH;
  const _Float16* K  = kh + (long)bh * LSEQ * DH;
  const _Float16* Vt = vt + (long)bh * DH * LSEQ;

  const int rb  = wave & 3;                 // C-tile row block for S / O
  const int cb0 = (wave >> 2) * 2;          // two C-tile col blocks
  const int rof = (lane < 16) ? 0 : 8;
  const int cc  = lane & 15;

#if USE_TDM
  const unsigned kp_lds_addr = (unsigned)(uintptr_t)(&KP_lds[0][0]);
  const unsigned vv_lds_addr = (unsigned)(uintptr_t)(&Ef_lds[0][0]);
#endif

  v8f o0 = {}; v8f o1 = {};
  float m_i = -1e30f, l_i = 0.0f;           // owned by tid<64 (row tid)

  for (int kt = 0; kt < KTILES; ++kt) {
    const int m0 = kt * 64;
    const int delta = m0 - l0;
    __syncthreads();

#if USE_TDM
    // ---- 0) TDM: DMA K tile (64 rows x 64 f16, row stride 64) into KP_lds ----
    if (wave == 0)
      tdm_load_tile_f16(kp_lds_addr, K + (long)m0 * DH, 64u, 64u, 64ul);
#endif

    // ---- 1) build folded-E window: row j <-> d = j+delta-63 ----
    {
      int j  = tid >> 1;
      int c0 = (tid & 1) * 32;
      int d  = j + delta - 63;
      if (d == 1 || j > 126) {
#pragma unroll
        for (int c = 0; c < 32; ++c) Ef_lds[j][c0 + c] = (_Float16)0.0f;
      } else {
        int e = (d <= 0) ? (LSEQ - 1 + d) : (d - 2);
        e = (e < 0) ? 0 : (e > LSEQ - 1 ? LSEQ - 1 : e);
        const float* er = E + (long)e * DH + c0;
#pragma unroll
        for (int c = 0; c < 32; c += 4) {
          v4f f = *(const v4f*)(er + c);
          Ef_lds[j][c0 + c + 0] = (_Float16)f[0];
          Ef_lds[j][c0 + c + 1] = (_Float16)f[1];
          Ef_lds[j][c0 + c + 2] = (_Float16)f[2];
          Ef_lds[j][c0 + c + 3] = (_Float16)f[3];
        }
      }
    }

#if USE_TDM
    if (wave == 0) __builtin_amdgcn_s_wait_tensorcnt(0);
#endif
    __syncthreads();   // Ef + K tile ready

    // ---- 2) S = Q K^T (two 16x16 C tiles per wave, K-dim = 64) ----
    {
      v8f s0 = {}; v8f s1 = {};
#pragma unroll
      for (int kk = 0; kk < 2; ++kk) {
        v16h a = frag_a(Q, DH, l0 + rb * 16 + cc, kk * 32, lane);
#if USE_TDM
        v16h b0 = frag_b(&KP_lds[0][0], 64, cb0 * 16 + cc, kk * 32, lane);
        v16h b1 = frag_b(&KP_lds[0][0], 64, (cb0 + 1) * 16 + cc, kk * 32, lane);
#else
        v16h b0 = frag_b(K, DH, m0 + cb0 * 16 + cc, kk * 32, lane);
        v16h b1 = frag_b(K, DH, m0 + (cb0 + 1) * 16 + cc, kk * 32, lane);
#endif
        s0 = WMMA_F16(a, b0, s0);
        s1 = WMMA_F16(a, b1, s1);
      }
#pragma unroll
      for (int r = 0; r < 8; ++r) {
        S_lds[rb * 16 + r + rof][cb0 * 16 + cc]       = s0[r];
        S_lds[rb * 16 + r + rof][(cb0 + 1) * 16 + cc] = s1[r];
      }
    }

    // ---- 3) G = Q[l0..l0+79] · Efold^T (wave = one 16-col j block) ----
    {
      const int jb = wave;
      v8f g[5];
#pragma unroll
      for (int i = 0; i < 5; ++i) g[i] = (v8f){};
#pragma unroll
      for (int kk = 0; kk < 2; ++kk) {
        v16h bf = frag_b(&Ef_lds[0][0], DH, jb * 16 + cc, kk * 32, lane);
#pragma unroll
        for (int rbk = 0; rbk < 5; ++rbk) {
          int row = l0 + rbk * 16 + cc;
          row = (row < LSEQ - 1) ? row : (LSEQ - 1);   // clamp tail reads
          v16h a = frag_a(Q, DH, row, kk * 32, lane);
          g[rbk] = WMMA_F16(a, bf, g[rbk]);
        }
      }
#pragma unroll
      for (int rbk = 0; rbk < 5; ++rbk)
#pragma unroll
        for (int r = 0; r < 8; ++r) {
          int gr = rbk * 16 + r + rof;
          if (gr < 66) G_lds[gr][jb * 16 + cc] = (_Float16)g[rbk][r];
        }
    }

    __syncthreads();   // S/G ready; K reads done (KP free), Ef reads done (V region free)

#if USE_TDM
    // ---- 3b) TDM: DMA V^T tile (64 d-rows x 64 f16, row stride LSEQ);
    //          overlaps the softmax pass below ----
    if (wave == 0)
      tdm_load_tile_f16(vv_lds_addr, Vt + m0, 64u, 64u, (unsigned long)LSEQ);
#endif

    // ---- 4) online softmax; faithful skew gather.  d = delta+mm-lr:
    //        d<=0 -> G[lr], d==1 -> 0 (zero Ef row), d>=2 -> G[lr+1]. ----
    if (tid < 64) {
      const int lr = tid;
      float mx = m_i;
#pragma unroll 8
      for (int mm = 0; mm < 64; ++mm) {
        int j = mm - lr + 63;
        int rsel = lr + ((delta + mm - lr) >= 1 ? 1 : 0);
        float s = S_lds[lr][mm] + (float)G_lds[rsel][j];
        mx = fmaxf(mx, s);
      }
      float alpha = __expf(m_i - mx);
      float sum = 0.0f;
#pragma unroll 8
      for (int mm = 0; mm < 64; ++mm) {
        int j = mm - lr + 63;
        int rsel = lr + ((delta + mm - lr) >= 1 ? 1 : 0);
        float s = S_lds[lr][mm] + (float)G_lds[rsel][j];
        float p = __expf(s - mx);
        sum += p;
        KP_lds[lr][mm] = (_Float16)p;       // P tile reuses the K-tile buffer
      }
      l_i = l_i * alpha + sum;
      m_i = mx;
      alpha_lds[lr] = alpha;
    }

#if USE_TDM
    if (wave == 0) __builtin_amdgcn_s_wait_tensorcnt(0);
#endif
    __syncthreads();   // P, alpha, V tile ready

    // ---- 5) rescale O, O += P · V ----
    {
#pragma unroll
      for (int r = 0; r < 8; ++r) {
        float a = alpha_lds[rb * 16 + r + rof];
        o0[r] *= a;
        o1[r] *= a;
      }
#pragma unroll
      for (int kk = 0; kk < 2; ++kk) {
        v16h a = frag_a(&KP_lds[0][0], 64, rb * 16 + cc, kk * 32, lane);
#if USE_TDM
        v16h b0 = frag_b(&Ef_lds[0][0], 64, cb0 * 16 + cc, kk * 32, lane);
        v16h b1 = frag_b(&Ef_lds[0][0], 64, (cb0 + 1) * 16 + cc, kk * 32, lane);
#else
        v16h b0 = frag_b(Vt, LSEQ, cb0 * 16 + cc, m0 + kk * 32, lane);
        v16h b1 = frag_b(Vt, LSEQ, (cb0 + 1) * 16 + cc, m0 + kk * 32, lane);
#endif
        o0 = WMMA_F16(a, b0, o0);
        o1 = WMMA_F16(a, b1, o1);
      }
      if (kt + 1 < KTILES) {   // warm L2 for the next tiles -> global_prefetch_b8
        __builtin_prefetch(K + ((long)(m0 + 64) + (lane << 1)) * DH, 0, 1);
        __builtin_prefetch(Vt + (long)(cb0 * 16 + cc) * LSEQ + m0 + 64, 0, 1);
      }
    }
  }

  // ---- epilogue: out = O / (rowsum * sqrt(dh)), sqrt(64)=8 (div AFTER softmax) ----
  __syncthreads();
  if (tid < 64) scale_lds[tid] = 1.0f / (l_i * 8.0f);
  __syncthreads();
  {
    float* outp = out + (long)b * LSEQ * DMODEL + (long)h * DH;
#pragma unroll
    for (int r = 0; r < 8; ++r) {
      int lr = rb * 16 + r + rof;
      float sc = scale_lds[lr];
      long rowoff = (long)(l0 + lr) * DMODEL;
      outp[rowoff + cb0 * 16 + cc]        = o0[r] * sc;
      outp[rowoff + (cb0 + 1) * 16 + cc]  = o1[r] * sc;
    }
  }
}

// ---------------------------------------------------------------------------
extern "C" void kernel_launch(void* const* d_in, const int* in_sizes, int n_in,
                              void* d_out, int out_size, void* d_ws, size_t ws_size,
                              hipStream_t stream) {
  const float* q  = (const float*)d_in[0];
  const float* k  = (const float*)d_in[1];
  const float* v  = (const float*)d_in[2];
  const float* Wq = (const float*)d_in[3];
  const float* Wk = (const float*)d_in[4];
  const float* Wv = (const float*)d_in[5];
  const float* E  = (const float*)d_in[6];

  const int B = in_sizes[0] / (LSEQ * DMODEL);          // 4
  const size_t perBuf = (size_t)B * NH * LSEQ * DH;     // halves per projected tensor

  _Float16* qh = (_Float16*)d_ws;                       // 8 MB
  _Float16* kh = qh + perBuf;                           // 8 MB
  _Float16* vt = kh + perBuf;                           // 8 MB (transposed V)

  const int mtiles = (B * LSEQ) / 64;                   // 128
  rga_proj<<<dim3(3 * mtiles * (DMODEL / 64)), dim3(256), 0, stream>>>(
      q, k, v, Wq, Wk, Wv, qh, kh, vt);

  rga_attn<<<dim3(B * NH * KTILES), dim3(256), 0, stream>>>(
      qh, kh, vt, E, (float*)d_out);
}
